// Hands4D_86715389706287
// MI455X (gfx1250) — compile-verified, matrix operand order checked by reading, not run on hardware
//
#include <hip/hip_runtime.h>
#include <math.h>

typedef _Float16 f16;
typedef _Float16 h8v  __attribute__((ext_vector_type(8)));
typedef _Float16 h16v __attribute__((ext_vector_type(16)));
typedef float    f8v  __attribute__((ext_vector_type(8)));

static constexpr int cB  = 8,  cT = 512, cJ = 21, cF = 512, cD = 256;
static constexpr int cNH = 4,  cHS = 64;
static constexpr int cN  = cB * cT;        // 4096
static constexpr int cNJ = cN * cJ;        // 86016
static constexpr int cH  = 128;            // LSTM hidden per direction
static constexpr int cG  = 4 * cH;         // 512 gates per direction

// ---------------------------------------------------------------------------
// Generic WMMA GEMM: C[M,N](f32) = A[M,K](f16,row) * B[K,N](f16,row)
// epilogue: + bias[n]; BN(n); act(1=relu). M%128==0, K%32==0; N arbitrary.
// Block = 128 threads (4 waves). 128x64 C tile; wave w owns rows [32w,32w+32):
// 2 A-fragments x 4 B-fragments = 8 WMMAs per 32-K step (B frags reused).
// A tile staged with GLOBAL_LOAD_ASYNC_TO_LDS_B128 (ASYNCcnt DMA path);
// B tile staged synchronously with an in-LDS transpose so fragment loads are
// contiguous ds_load_b128 pairs.
// ---------------------------------------------------------------------------
#define GLDA 40
#define GLDB 40

__global__ __launch_bounds__(128) void k_gemm_f16(
    const f16* __restrict__ A, const f16* __restrict__ B, float* __restrict__ C,
    int M, int N, int K,
    const float* __restrict__ bias,
    const float* __restrict__ bng, const float* __restrict__ bnb,
    const float* __restrict__ bnm, const float* __restrict__ bnv,
    int act)
{
  __shared__ __align__(16) f16 As[128 * GLDA];
  __shared__ __align__(16) f16 Bs[64 * GLDB];   // transposed: Bs[n_local][k]
  const int tid  = threadIdx.x;
  const int wave = tid >> 5;
  const int lane = tid & 31;
  const int lh   = lane & 15;
  const int kb   = (lane < 16) ? 0 : 8;
  const int mtile = blockIdx.y << 7;
  const int ntile = blockIdx.x << 6;
  const bool nvec = ((N & 7) == 0) && (ntile + 64 <= N);

  f8v acc[2][4] = {};

  for (int k0 = 0; k0 < K; k0 += 32) {
    // ---- A tile 128x32: async DMA straight into LDS (16B per lane-op)
#pragma unroll
    for (int it = 0; it < 4; ++it) {
      int c = tid + (it << 7);            // 0..511 chunks of 8 halves
      int row = c >> 2, col = (c & 3) << 3;
      unsigned long long ga = (unsigned long long)(const void*)(A + (size_t)(mtile + row) * K + (k0 + col));
      unsigned int       la = (unsigned int)(unsigned long long)(const void*)(As + row * GLDA + col);
      asm volatile("global_load_async_to_lds_b128 %0, %1, off"
                   :: "v"(la), "v"(ga) : "memory");
    }
    // ---- B tile 32x64, stored transposed so frag loads are contiguous
#pragma unroll
    for (int it = 0; it < 2; ++it) {
      int c = tid + (it << 7);            // 0..255
      int kr = c >> 3, nc = (c & 7) << 3;
      f16 tmp[8];
      if (nvec) {
        h8v v = *(const h8v*)(B + (size_t)(k0 + kr) * N + (ntile + nc));
#pragma unroll
        for (int q = 0; q < 8; ++q) tmp[q] = v[q];
      } else {
#pragma unroll
        for (int q = 0; q < 8; ++q) {
          int n = ntile + nc + q;
          tmp[q] = (n < N) ? B[(size_t)(k0 + kr) * N + n] : (f16)0.0f;
        }
      }
#pragma unroll
      for (int q = 0; q < 8; ++q) Bs[(nc + q) * GLDB + kr] = tmp[q];
    }
    if (k0 + 32 < K) {   // global_prefetch_b8 of next B K-tile
      __builtin_prefetch(B + (size_t)(k0 + 32 + (tid & 31)) * N + ntile, 0, 1);
    }
    asm volatile("s_wait_asynccnt 0" ::: "memory");
    __syncthreads();

    // A fragments: lane holds row m=lh, K = kb+0..7 and kb+16..23
    h16v af[2];
#pragma unroll
    for (int mi = 0; mi < 2; ++mi) {
      const f16* ap = As + ((wave << 5) + (mi << 4) + lh) * GLDA + kb;
#pragma unroll
      for (int q = 0; q < 8; ++q) { af[mi][q] = ap[q]; af[mi][8 + q] = ap[16 + q]; }
    }
    // B fragments (column n = nt*16+lh), same K striping; reused for 2 A frags
#pragma unroll
    for (int nt = 0; nt < 4; ++nt) {
      h16v bf;
      const f16* bp = Bs + ((nt << 4) + lh) * GLDB + kb;
#pragma unroll
      for (int q = 0; q < 8; ++q) { bf[q] = bp[q]; bf[8 + q] = bp[16 + q]; }
      acc[0][nt] = __builtin_amdgcn_wmma_f32_16x16x32_f16(false, af[0], false, bf, (short)0, acc[0][nt], false, false);
      acc[1][nt] = __builtin_amdgcn_wmma_f32_16x16x32_f16(false, af[1], false, bf, (short)0, acc[1][nt], false, false);
    }
    __syncthreads();
  }

  // D layout: lane n=lh; VGPR r -> row r + (lane<16 ? 0 : 8)
#pragma unroll
  for (int mi = 0; mi < 2; ++mi) {
    const int mrow = mtile + (wave << 5) + (mi << 4) + ((lane < 16) ? 0 : 8);
#pragma unroll
    for (int nt = 0; nt < 4; ++nt) {
      int n = ntile + (nt << 4) + lh;
      if (n >= N) continue;
      float bval = bias ? bias[n] : 0.0f;
      float sc = 1.0f, sf = 0.0f;
      if (bng) { float inv = rsqrtf(bnv[n] + 1e-5f); sc = bng[n] * inv; sf = bnb[n] - bnm[n] * sc; }
#pragma unroll
      for (int r = 0; r < 8; ++r) {
        float y = acc[mi][nt][r] + bval;
        y = y * sc + sf;
        if (act == 1) y = fmaxf(y, 0.0f);
        C[(size_t)(mrow + r) * N + n] = y;
      }
    }
  }
}

// ---------------------------------------------------------------------------
// Elementwise / packing kernels
// ---------------------------------------------------------------------------
__global__ void k_f32_to_f16(const float* __restrict__ s, f16* __restrict__ d, int n) {
  int i = blockIdx.x * blockDim.x + threadIdx.x;
  if (i < n) d[i] = (f16)s[i];
}

// W [4][Fin][64] f32 -> out [Fin][256] f16  (head-concat columns)
__global__ void k_pack_gat_w(const float* __restrict__ W, f16* __restrict__ out, int Fin) {
  int i = blockIdx.x * blockDim.x + threadIdx.x;
  int tot = cNH * Fin * cHS;
  if (i >= tot) return;
  int d = i % cHS; int rest = i / cHS; int f = rest % Fin; int h = rest / Fin;
  out[(size_t)f * (cNH * cHS) + h * cHS + d] = (f16)W[i];
}

// w [O][I] f32 -> out[i*ldo + colofs + o] f16 (transpose pack)
__global__ void k_pack_transpose(const float* __restrict__ w, f16* __restrict__ out,
                                 int O, int I, int ldo, int colofs) {
  int i = blockIdx.x * blockDim.x + threadIdx.x;
  if (i >= O * I) return;
  int ii = i % I; int o = i / I;
  out[(size_t)ii * ldo + colofs + o] = (f16)w[i];
}

// conv w [O=256][I=256][k=3] -> out[(k*256+i)][o] f16
__global__ void k_pack_conv_w(const float* __restrict__ w, f16* __restrict__ out) {
  int i = blockIdx.x * blockDim.x + threadIdx.x;
  int tot = cD * cD * 3;
  if (i >= tot) return;
  int k = i % 3; int rest = i / 3; int ci = rest % cD; int o = rest / cD;
  out[(size_t)(k * cD + ci) * cD + o] = (f16)w[i];
}

// combined LSTM bias: [0..511]=bih_f+bhh_f, [512..1023]=bih_b+bhh_b
__global__ void k_lstm_bias(const float* __restrict__ a1, const float* __restrict__ a2,
                            const float* __restrict__ b1, const float* __restrict__ b2,
                            float* __restrict__ out) {
  int i = blockIdx.x * blockDim.x + threadIdx.x;
  if (i < cG) out[i] = a1[i] + a2[i];
  else if (i < 2 * cG) out[i] = b1[i - cG] + b2[i - cG];
}

// s1/s2[row*4+h] = <Wh[row, h*64 .. +63], a[h, 0..63 / 64..127]>
__global__ void k_gat_scores(const float* __restrict__ Wh, const float* __restrict__ a,
                             float* __restrict__ s1, float* __restrict__ s2) {
  int i = blockIdx.x * blockDim.x + threadIdx.x;
  if (i >= cNJ * cNH) return;
  int h = i & 3; int row = i >> 2;
  const float* wp = Wh + (size_t)row * cD + h * cHS;
  const float* a1 = a + h * (2 * cHS);
  const float* a2 = a1 + cHS;
  float x1 = 0.0f, x2 = 0.0f;
  for (int d = 0; d < cHS; ++d) { float v = wp[d]; x1 += v * a1[d]; x2 += v * a2[d]; }
  s1[i] = x1; s2[i] = x2;
}

// per (bt, head): leakyrelu scores, adj mask, softmax, aggregate, ELU
__global__ __launch_bounds__(64) void k_gat_attn(
    const float* __restrict__ Wh, const float* __restrict__ s1,
    const float* __restrict__ s2, const float* __restrict__ adj,
    float* __restrict__ out)
{
  __shared__ float att[cJ * cJ];
  __shared__ float sh1[cJ], sh2[cJ];
  __shared__ float wt[cJ * cHS];
  int bt = blockIdx.x >> 2;
  int h  = blockIdx.x & 3;
  int tid = threadIdx.x;
  if (tid < cJ) {
    sh1[tid] = s1[((size_t)bt * cJ + tid) * cNH + h];
    sh2[tid] = s2[((size_t)bt * cJ + tid) * cNH + h];
  }
  for (int p = tid; p < cJ * cHS; p += 64) {
    int j = p / cHS, d = p % cHS;
    wt[p] = Wh[((size_t)bt * cJ + j) * cD + h * cHS + d];
  }
  __syncthreads();
  for (int p = tid; p < cJ * cJ; p += 64) {
    int i = p / cJ, j = p % cJ;
    float e = sh1[i] + sh2[j];
    e = (e > 0.0f) ? e : 0.2f * e;
    att[p] = (adj[p] > 0.0f) ? e : -9e15f;
  }
  __syncthreads();
  if (tid < cJ) {
    float mx = -3.4e38f;
    for (int j = 0; j < cJ; ++j) mx = fmaxf(mx, att[tid * cJ + j]);
    float sum = 0.0f;
    for (int j = 0; j < cJ; ++j) { float v = __expf(att[tid * cJ + j] - mx); att[tid * cJ + j] = v; sum += v; }
    float inv = 1.0f / sum;
    for (int j = 0; j < cJ; ++j) att[tid * cJ + j] *= inv;
  }
  __syncthreads();
  int d = tid;   // 64 threads = 64 feature dims
  for (int i = 0; i < cJ; ++i) {
    float acc = 0.0f;
    for (int j = 0; j < cJ; ++j) acc += att[i * cJ + j] * wt[j * cHS + d];
    acc = (acc > 0.0f) ? acc : (__expf(acc) - 1.0f);   // ELU
    out[((size_t)bt * cJ + i) * cD + h * cHS + d] = acc;
  }
}

// MHA core per (n, head), L=21, hd=64; writes output in-place over Q
__global__ __launch_bounds__(64) void k_mha_attn(
    float* __restrict__ Q, const float* __restrict__ K, const float* __restrict__ V)
{
  __shared__ float qt[cJ * cHS], kt[cJ * cHS], vt[cJ * cHS];
  __shared__ float att[cJ * cJ];
  int n = blockIdx.x >> 2;
  int h = blockIdx.x & 3;
  int tid = threadIdx.x;
  for (int p = tid; p < cJ * cHS; p += 64) {
    int l = p / cHS, d = p % cHS;
    size_t idx = ((size_t)n * cJ + l) * cD + h * cHS + d;
    qt[p] = Q[idx]; kt[p] = K[idx]; vt[p] = V[idx];
  }
  __syncthreads();
  for (int p = tid; p < cJ * cJ; p += 64) {
    int i = p / cJ, j = p % cJ;
    float s = 0.0f;
    for (int d = 0; d < cHS; ++d) s += qt[i * cHS + d] * kt[j * cHS + d];
    att[p] = s * 0.125f;
  }
  __syncthreads();
  if (tid < cJ) {
    float mx = -3.4e38f;
    for (int j = 0; j < cJ; ++j) mx = fmaxf(mx, att[tid * cJ + j]);
    float sum = 0.0f;
    for (int j = 0; j < cJ; ++j) { float v = __expf(att[tid * cJ + j] - mx); att[tid * cJ + j] = v; sum += v; }
    float inv = 1.0f / sum;
    for (int j = 0; j < cJ; ++j) att[tid * cJ + j] *= inv;
  }
  __syncthreads();
  int d = tid;
  for (int i = 0; i < cJ; ++i) {
    float acc = 0.0f;
    for (int j = 0; j < cJ; ++j) acc += att[i * cJ + j] * vt[j * cHS + d];
    Q[((size_t)n * cJ + i) * cD + h * cHS + d] = acc;
  }
}

// GAT output [(b*T+t)*J+j, d] f32 -> im2col f16 rows (b*J+j)*T+t, cols k*256+i
__global__ void k_im2col_gat(const float* __restrict__ g, f16* __restrict__ out, int ntot) {
  int i = blockIdx.x * blockDim.x + threadIdx.x;
  if (i >= ntot) return;
  int c = i % (3 * cD); int r = i / (3 * cD);
  int k = c / cD, ch = c % cD;
  int t = r % cT; int bj = r / cT;
  int b = bj / cJ, j = bj % cJ;
  int ts = t + k - 1;
  float v = 0.0f;
  if (ts >= 0 && ts < cT) v = g[(((size_t)b * cT + ts) * cJ + j) * cD + ch];
  out[i] = (f16)v;
}

// y [(bj)*T+t, d] f32 -> im2col f16 (same row layout)
__global__ void k_im2col_seq(const float* __restrict__ y, f16* __restrict__ out, int ntot) {
  int i = blockIdx.x * blockDim.x + threadIdx.x;
  if (i >= ntot) return;
  int c = i % (3 * cD); int r = i / (3 * cD);
  int k = c / cD, ch = c % cD;
  int t = r % cT; int bj = r / cT;
  int ts = t + k - 1;
  float v = 0.0f;
  if (ts >= 0 && ts < cT) v = y[((size_t)bj * cT + ts) * cD + ch];
  out[i] = (f16)v;
}

// conv output [(b*J+j)*T+t, d] -> sequence layout [(b*T+t)*J+j, d] (f32 + f16)
__global__ void k_conv_to_seq(const float* __restrict__ y, float* __restrict__ s,
                              f16* __restrict__ s16, int ntot) {
  int i = blockIdx.x * blockDim.x + threadIdx.x;
  if (i >= ntot) return;
  int d = i % cD; int n = i / cD;     // n = bt*J + j
  int j = n % cJ; int bt = n / cJ;
  int b = bt / cT, t = bt % cT;
  float v = y[(((size_t)b * cJ + j) * cT + t) * cD + d];
  s[i] = v; s16[i] = (f16)v;
}

// e[bt, d] = mean_j (base + 0.5*attn) ; also f16 copy
__global__ void k_combine_mean(const float* __restrict__ base, const float* __restrict__ attn,
                               float* __restrict__ e, f16* __restrict__ e16, int ntot) {
  int i = blockIdx.x * blockDim.x + threadIdx.x;
  if (i >= ntot) return;
  int d = i % cD; int bt = i / cD;
  float acc = 0.0f;
  for (int j = 0; j < cJ; ++j) {
    size_t idx = ((size_t)bt * cJ + j) * cD + d;
    acc += base[idx] + 0.5f * attn[idx];
  }
  acc *= (1.0f / (float)cJ);
  e[i] = acc; e16[i] = (f16)acc;
}

// One block per direction; sequential over T with state in LDS.
// xw: [cN, 1024] (cols 0..511 fw gates, 512..1023 bw); out: [cN, 256]
__global__ __launch_bounds__(512) void k_lstm(
    const float* __restrict__ xw,
    const float* __restrict__ whh_f, const float* __restrict__ whh_b,
    float* __restrict__ out)
{
  __shared__ float hst[cB * cH];
  __shared__ float cst[cB * cH];
  __shared__ float gst[cB * cG];
  const int dir = blockIdx.x;
  const float* whh = dir ? whh_b : whh_f;
  const int co = dir * cG;
  const int oo = dir * cH;
  const int tid = threadIdx.x;
  for (int i = tid; i < cB * cH; i += 512) { hst[i] = 0.0f; cst[i] = 0.0f; }
  __syncthreads();
  for (int step = 0; step < cT; ++step) {
    int t = dir ? (cT - 1 - step) : step;
#pragma unroll
    for (int q = 0; q < 8; ++q) {
      int gi = tid + q * 512;              // 0..4095
      int b = gi >> 9, u = gi & 511;
      const float* hp = hst + b * cH;
      const float* wp = whh + (size_t)u * cH;
      float acc = xw[((size_t)b * cT + t) * (2 * cG) + co + u];
      for (int kk = 0; kk < cH; ++kk) acc += hp[kk] * wp[kk];
      gst[gi] = acc;
    }
    __syncthreads();
#pragma unroll
    for (int q = 0; q < 2; ++q) {
      int ci = tid + q * 512;              // 0..1023
      int b = ci >> 7, u = ci & 127;
      float gi = gst[b * cG + u];
      float gf = gst[b * cG + cH + u];
      float gg = gst[b * cG + 2 * cH + u];
      float go = gst[b * cG + 3 * cH + u];
      float si = 1.0f / (1.0f + __expf(-gi));
      float sf = 1.0f / (1.0f + __expf(-gf));
      float so = 1.0f / (1.0f + __expf(-go));
      float cc = sf * cst[ci] + si * tanhf(gg);
      float hh = so * tanhf(cc);
      cst[ci] = cc; hst[ci] = hh;
      out[((size_t)b * cT + t) * (2 * cH) + oo + u] = hh;
    }
    __syncthreads();
  }
}

__global__ void k_finalize(const float* __restrict__ y, const float* __restrict__ vis,
                           float* __restrict__ out, int ntot) {
  int i = blockIdx.x * blockDim.x + threadIdx.x;
  if (i >= ntot) return;
  int q = i % 63; int n = i / 63;
  int j = q / 3;
  out[i] = y[i] * vis[(size_t)n * cJ + j];
}

// ---------------------------------------------------------------------------
// Host orchestration
// ---------------------------------------------------------------------------
#define LAUNCH1D(kern, n, ...) \
  kern<<<dim3((unsigned)(((n) + 255) / 256)), dim3(256), 0, stream>>>(__VA_ARGS__)

extern "C" void kernel_launch(void* const* d_in, const int* in_sizes, int n_in,
                              void* d_out, int out_size, void* d_ws, size_t ws_size,
                              hipStream_t stream) {
  (void)in_sizes; (void)n_in; (void)out_size; (void)ws_size;
  const float* lh  = (const float*)d_in[0];
  const float* rh  = (const float*)d_in[1];
  const float* adj = (const float*)d_in[2];
  const float* vis = (const float*)d_in[3];
  const float* g1W = (const float*)d_in[4];
  const float* g1a = (const float*)d_in[5];
  const float* g2W = (const float*)d_in[6];
  const float* g2a = (const float*)d_in[7];
  const float* c1w = (const float*)d_in[8];
  const float* c1b = (const float*)d_in[9];
  const float* c2w = (const float*)d_in[10];
  const float* c2b = (const float*)d_in[11];
  const float* b1g = (const float*)d_in[12];
  const float* b1b = (const float*)d_in[13];
  const float* b1m = (const float*)d_in[14];
  const float* b1v = (const float*)d_in[15];
  const float* b2g = (const float*)d_in[16];
  const float* b2b = (const float*)d_in[17];
  const float* b2m = (const float*)d_in[18];
  const float* b2v = (const float*)d_in[19];
  const float* wq  = (const float*)d_in[20];
  const float* wk  = (const float*)d_in[21];
  const float* wv  = (const float*)d_in[22];
  const float* wo  = (const float*)d_in[23];
  const float* bq  = (const float*)d_in[24];
  const float* bk  = (const float*)d_in[25];
  const float* bv  = (const float*)d_in[26];
  const float* bo  = (const float*)d_in[27];
  const float* l0f_wih = (const float*)d_in[28];
  const float* l0f_whh = (const float*)d_in[29];
  const float* l0f_bih = (const float*)d_in[30];
  const float* l0f_bhh = (const float*)d_in[31];
  const float* l0b_wih = (const float*)d_in[32];
  const float* l0b_whh = (const float*)d_in[33];
  const float* l0b_bih = (const float*)d_in[34];
  const float* l0b_bhh = (const float*)d_in[35];
  const float* l1f_wih = (const float*)d_in[36];
  const float* l1f_whh = (const float*)d_in[37];
  const float* l1f_bih = (const float*)d_in[38];
  const float* l1f_bhh = (const float*)d_in[39];
  const float* l1b_wih = (const float*)d_in[40];
  const float* l1b_whh = (const float*)d_in[41];
  const float* l1b_bih = (const float*)d_in[42];
  const float* l1b_bhh = (const float*)d_in[43];
  const float* o1w = (const float*)d_in[44];
  const float* o1b = (const float*)d_in[45];
  const float* o2w = (const float*)d_in[46];
  const float* o2b = (const float*)d_in[47];

  // ---- workspace carve (deterministic arithmetic only)
  size_t off = 0;
  auto carve = [&](size_t bytes) { size_t o = off; off += (bytes + 255) & ~(size_t)255; return o; };
  char* base = (char*)d_ws;
  const size_t oA16  = carve((size_t)cNJ * 768 * 2);    // f16 GEMM-A staging (largest: im2col)
  const size_t oC0   = carve((size_t)cNJ * cD * 4);
  const size_t oC1   = carve((size_t)cNJ * cD * 4);
  const size_t oC2   = carve((size_t)cNJ * cD * 4);
  const size_t oLF   = carve((size_t)cNJ * cD * 4);
  const size_t oRF   = carve((size_t)cNJ * cD * 4);
  const size_t oLF16 = carve((size_t)cNJ * cD * 2);
  const size_t oRF16 = carve((size_t)cNJ * cD * 2);
  const size_t oS1   = carve((size_t)cNJ * cNH * 4);
  const size_t oS2   = carve((size_t)cNJ * cNH * 4);
  const size_t oGW1  = carve((size_t)cF * cD * 2);
  const size_t oGW2  = carve((size_t)cD * cD * 2);
  const size_t oCW1  = carve((size_t)3 * cD * cD * 2);
  const size_t oCW2  = carve((size_t)3 * cD * cD * 2);
  const size_t oWQ   = carve((size_t)cD * cD * 2);
  const size_t oWK   = carve((size_t)cD * cD * 2);
  const size_t oWV   = carve((size_t)cD * cD * 2);
  const size_t oWO   = carve((size_t)cD * cD * 2);
  const size_t oLW0  = carve((size_t)cD * 1024 * 2);
  const size_t oLW1  = carve((size_t)cD * 1024 * 2);
  const size_t oLB0  = carve((size_t)1024 * 4);
  const size_t oLB1  = carve((size_t)1024 * 4);
  const size_t oHW1  = carve((size_t)cD * cD * 2);
  const size_t oHW2  = carve((size_t)cD * 63 * 2);
  const size_t oLE   = carve((size_t)cN * cD * 4);
  const size_t oRE   = carve((size_t)cN * cD * 4);
  const size_t oLE16 = carve((size_t)cN * cD * 2);
  const size_t oRE16 = carve((size_t)cN * cD * 2);
  const size_t oXW   = carve((size_t)cN * 1024 * 4);
  const size_t oH1   = carve((size_t)cN * cD * 4);
  const size_t oH2   = carve((size_t)cN * cD * 4);
  const size_t oHD1  = carve((size_t)cN * cD * 4);
  const size_t oY    = carve((size_t)cN * 63 * 4);

  f16*   A16  = (f16*)(base + oA16);
  float* C0   = (float*)(base + oC0);
  float* C1   = (float*)(base + oC1);
  float* C2   = (float*)(base + oC2);
  float* LF   = (float*)(base + oLF);
  float* RF   = (float*)(base + oRF);
  f16*   LF16 = (f16*)(base + oLF16);
  f16*   RF16 = (f16*)(base + oRF16);
  float* S1   = (float*)(base + oS1);
  float* S2   = (float*)(base + oS2);
  f16*   GW1  = (f16*)(base + oGW1);
  f16*   GW2  = (f16*)(base + oGW2);
  f16*   CW1  = (f16*)(base + oCW1);
  f16*   CW2  = (f16*)(base + oCW2);
  f16*   WQt  = (f16*)(base + oWQ);
  f16*   WKt  = (f16*)(base + oWK);
  f16*   WVt  = (f16*)(base + oWV);
  f16*   WOt  = (f16*)(base + oWO);
  f16*   LW0  = (f16*)(base + oLW0);
  f16*   LW1  = (f16*)(base + oLW1);
  float* LB0  = (float*)(base + oLB0);
  float* LB1  = (float*)(base + oLB1);
  f16*   HW1  = (f16*)(base + oHW1);
  f16*   HW2  = (f16*)(base + oHW2);
  float* LE   = (float*)(base + oLE);
  float* RE   = (float*)(base + oRE);
  f16*   LE16 = (f16*)(base + oLE16);
  f16*   RE16 = (f16*)(base + oRE16);
  float* XW   = (float*)(base + oXW);
  float* H1   = (float*)(base + oH1);
  float* H2   = (float*)(base + oH2);
  float* HD1  = (float*)(base + oHD1);
  float* Y    = (float*)(base + oY);

  auto gemm = [&](const f16* A, const f16* Bm, float* C, int M, int N, int K,
                  const float* bias, const float* bng, const float* bnb,
                  const float* bnm, const float* bnv, int act) {
    dim3 grid((unsigned)((N + 63) / 64), (unsigned)(M / 128));
    k_gemm_f16<<<grid, 128, 0, stream>>>(A, Bm, C, M, N, K, bias, bng, bnb, bnm, bnv, act);
  };

  // ---- pack weights (cheap, every call; deterministic)
  LAUNCH1D(k_pack_gat_w, cNH * cF * cHS, g1W, GW1, cF);
  LAUNCH1D(k_pack_gat_w, cNH * cD * cHS, g2W, GW2, cD);
  LAUNCH1D(k_pack_conv_w, cD * cD * 3, c1w, CW1);
  LAUNCH1D(k_pack_conv_w, cD * cD * 3, c2w, CW2);
  LAUNCH1D(k_pack_transpose, cD * cD, wq, WQt, cD, cD, cD, 0);
  LAUNCH1D(k_pack_transpose, cD * cD, wk, WKt, cD, cD, cD, 0);
  LAUNCH1D(k_pack_transpose, cD * cD, wv, WVt, cD, cD, cD, 0);
  LAUNCH1D(k_pack_transpose, cD * cD, wo, WOt, cD, cD, cD, 0);
  LAUNCH1D(k_pack_transpose, cG * cD, l0f_wih, LW0, cG, cD, 1024, 0);
  LAUNCH1D(k_pack_transpose, cG * cD, l0b_wih, LW0, cG, cD, 1024, cG);
  LAUNCH1D(k_pack_transpose, cG * cD, l1f_wih, LW1, cG, cD, 1024, 0);
  LAUNCH1D(k_pack_transpose, cG * cD, l1b_wih, LW1, cG, cD, 1024, cG);
  LAUNCH1D(k_lstm_bias, 2 * cG, l0f_bih, l0f_bhh, l0b_bih, l0b_bhh, LB0);
  LAUNCH1D(k_lstm_bias, 2 * cG, l1f_bih, l1f_bhh, l1b_bih, l1b_bhh, LB1);
  LAUNCH1D(k_pack_transpose, cD * cD, o1w, HW1, cD, cD, cD, 0);
  LAUNCH1D(k_pack_transpose, 63 * cD, o2w, HW2, 63, cD, 63, 0);

  // ---- per-hand front (GAT x2 + conv x2)
  for (int hand = 0; hand < 2; ++hand) {
    const float* x = hand ? rh : lh;
    float* seqF = hand ? RF : LF;
    f16*   seqH = hand ? RF16 : LF16;
    LAUNCH1D(k_f32_to_f16, cNJ * cF, x, A16, cNJ * cF);
    gemm(A16, GW1, C0, cNJ, cD, cF, nullptr, nullptr, nullptr, nullptr, nullptr, 0);
    LAUNCH1D(k_gat_scores, cNJ * cNH, C0, g1a, S1, S2);
    k_gat_attn<<<dim3(cN * cNH), dim3(64), 0, stream>>>(C0, S1, S2, adj, C1);
    LAUNCH1D(k_f32_to_f16, cNJ * cD, C1, A16, cNJ * cD);
    gemm(A16, GW2, C0, cNJ, cD, cD, nullptr, nullptr, nullptr, nullptr, nullptr, 0);
    LAUNCH1D(k_gat_scores, cNJ * cNH, C0, g2a, S1, S2);
    k_gat_attn<<<dim3(cN * cNH), dim3(64), 0, stream>>>(C0, S1, S2, adj, C1);
    LAUNCH1D(k_im2col_gat, cNJ * 3 * cD, C1, A16, cNJ * 3 * cD);
    gemm(A16, CW1, C0, cNJ, cD, 3 * cD, c1b, b1g, b1b, b1m, b1v, 1);
    LAUNCH1D(k_im2col_seq, cNJ * 3 * cD, C0, A16, cNJ * 3 * cD);
    gemm(A16, CW2, C1, cNJ, cD, 3 * cD, c2b, b2g, b2b, b2m, b2v, 1);
    LAUNCH1D(k_conv_to_seq, cNJ * cD, C1, seqF, seqH, cNJ * cD);
  }

  // ---- cross-hand MHA (dirn 0: a_l2r -> re ; dirn 1: a_r2l -> le)
  for (int dirn = 0; dirn < 2; ++dirn) {
    const f16* q16  = dirn == 0 ? LF16 : RF16;
    const f16* kv16 = dirn == 0 ? RF16 : LF16;
    const float* bf = dirn == 0 ? RF : LF;
    float* e   = dirn == 0 ? RE : LE;
    f16*   e16 = dirn == 0 ? RE16 : LE16;
    gemm(q16,  WQt, C0, cNJ, cD, cD, bq, nullptr, nullptr, nullptr, nullptr, 0);
    gemm(kv16, WKt, C1, cNJ, cD, cD, bk, nullptr, nullptr, nullptr, nullptr, 0);
    gemm(kv16, WVt, C2, cNJ, cD, cD, bv, nullptr, nullptr, nullptr, nullptr, 0);
    k_mha_attn<<<dim3(cN * cNH), dim3(64), 0, stream>>>(C0, C1, C2);
    LAUNCH1D(k_f32_to_f16, cNJ * cD, C0, A16, cNJ * cD);
    gemm(A16, WOt, C1, cNJ, cD, cD, bo, nullptr, nullptr, nullptr, nullptr, 0);
    LAUNCH1D(k_combine_mean, cN * cD, bf, C1, e, e16, cN * cD);
  }

  // ---- per-hand bi-LSTM x2 + head
  for (int hand = 0; hand < 2; ++hand) {
    f16* e16 = hand ? RE16 : LE16;
    float* outp = (float*)d_out + (size_t)hand * cN * 63;
    gemm(e16, LW0, XW, cN, 1024, cD, LB0, nullptr, nullptr, nullptr, nullptr, 0);
    k_lstm<<<dim3(2), dim3(512), 0, stream>>>(XW, l0f_whh, l0b_whh, H1);
    LAUNCH1D(k_f32_to_f16, cN * cD, H1, A16, cN * cD);
    gemm(A16, LW1, XW, cN, 1024, cD, LB1, nullptr, nullptr, nullptr, nullptr, 0);
    k_lstm<<<dim3(2), dim3(512), 0, stream>>>(XW, l1f_whh, l1b_whh, H2);
    LAUNCH1D(k_f32_to_f16, cN * cD, H2, A16, cN * cD);
    gemm(A16, HW1, HD1, cN, cD, cD, o1b, nullptr, nullptr, nullptr, nullptr, 1);
    LAUNCH1D(k_f32_to_f16, cN * cD, HD1, A16, cN * cD);
    gemm(A16, HW2, Y, cN, 63, cD, o2b, nullptr, nullptr, nullptr, nullptr, 0);
    LAUNCH1D(k_finalize, cN * 63, Y, vis, outp, cN * 63);
  }
}
